// SymEqNet_67997922230969
// MI455X (gfx1250) — matrix operand order repaired and checked by reading, not compiled, not run on hardware
//
#include <hip/hip_runtime.h>
#include <math.h>

#define N_NODES 50000
#define N_EDGES 800000
#define N_GRAPH 64
// F=64, 3F=192, 13F=832, HG=HL=128

typedef __attribute__((ext_vector_type(16))) _Float16 v16h;
typedef __attribute__((ext_vector_type(8)))  float    v8f;

// ---------------------------------------------------------------- helpers ---
__device__ __forceinline__ v8f wmma16(v16h a, v16h b, v8f c) {
  // D = A(16x32 f16) * B(32x16 f16) + C(16x16 f32)
  return __builtin_amdgcn_wmma_f32_16x16x32_f16(
      /*neg_a=*/false, a, /*neg_b=*/false, b,
      /*c_mod=*/(short)0, c, /*reuse_a=*/false, /*reuse_b=*/false);
}

// Monotone float<->ordered-int bijection (involution) for atomic min/max.
__device__ __forceinline__ int f2ord(float f) {
  int i = __float_as_int(f);
  return i >= 0 ? i : (i ^ 0x7fffffff);
}
__device__ __forceinline__ float ord2f(int i) {
  return __int_as_float(i >= 0 ? i : (i ^ 0x7fffffff));
}

__device__ __forceinline__ v8f v8f_zero() {
  v8f z = {0.f, 0.f, 0.f, 0.f, 0.f, 0.f, 0.f, 0.f};
  return z;
}

// ------------------------------------------------------------ init kernels --
__global__ void k_fill_f32(float* p, long n, float v) {
  for (long i = (long)blockIdx.x * blockDim.x + threadIdx.x; i < n;
       i += (long)gridDim.x * blockDim.x)
    p[i] = v;
}
__global__ void k_fill_i32(int* p, long n, int v) {
  for (long i = (long)blockIdx.x * blockDim.x + threadIdx.x; i < n;
       i += (long)gridDim.x * blockDim.x)
    p[i] = v;
}

// Convert a row-major f32 weight [KT*32, NT*16] (ldw cols) into WMMA
// B-fragment-ordered f16: out[((kk*NT+nt)*32 + lane)*16 + j]
//   = W[(kk*32 + K(lane,j)) * ldw + nt*16 + (lane&15)],  K = (j&7) + 8*(lane>=16) + 16*(j>=8)
__global__ void k_convert_frag(const float* __restrict__ W,
                               _Float16* __restrict__ out,
                               int KT, int NT, int ldw) {
  long total = (long)KT * NT * 512;
  for (long idx = (long)blockIdx.x * blockDim.x + threadIdx.x; idx < total;
       idx += (long)gridDim.x * blockDim.x) {
    int j    = (int)(idx & 15);
    int lane = (int)((idx >> 4) & 31);
    long fi  = idx >> 9;                 // fragment index = kk*NT + nt
    int nt   = (int)(fi % NT);
    int kk   = (int)(fi / NT);
    int K    = (j & 7) + ((lane >> 4) << 3) + ((j >> 3) << 4);
    out[idx] = (_Float16)W[(long)(kk * 32 + K) * ldw + nt * 16 + (lane & 15)];
  }
}

// ----------------------------------------------------------- count kernels --
__global__ void k_edge_count(const int* __restrict__ ei, float* __restrict__ cnt) {
  int e = blockIdx.x * blockDim.x + threadIdx.x;
  if (e < N_EDGES) atomicAdd(&cnt[ei[N_EDGES + e]], 1.0f);  // dst = ei[1][e]
}
__global__ void k_graph_count(const int* __restrict__ batch, float* __restrict__ gcnt) {
  int i = blockIdx.x * blockDim.x + threadIdx.x;
  if (i < N_NODES) atomicAdd(&gcnt[batch[i]], 1.0f);
}

// ------------------------------------------------------- edge GEMM + scatter -
// Per wave: 16 edges.  A = [16 x 192] f16 in LDS (x[dst] | x[src] | e),
// B = W_pre fragments (f16, fragment-ordered), C = [16 x 64] f32.
// Epilogue scatters into per-node sum / sum^2 / ordered-min / ordered-max
// (these 51 MB buffers are L2-resident on MI455X's 192 MB L2).
__global__ __launch_bounds__(256)
void k_edge_gemm(const float* __restrict__ x, const int* __restrict__ ei,
                 const float* __restrict__ ea, const float* __restrict__ Wedge,
                 const float* __restrict__ bedge, const float* __restrict__ bpre,
                 const _Float16* __restrict__ wpreF,
                 float* __restrict__ sum, float* __restrict__ sum2,
                 int* __restrict__ mink, int* __restrict__ maxk) {
  __shared__ _Float16 ldsA[8 * 16 * 192] __attribute__((aligned(16)));
  __shared__ int sdst[8 * 16];
  __shared__ int ssrc[8 * 16];

  const int lane = threadIdx.x & 31;
  const int wave = threadIdx.x >> 5;
  const int tileBase = (blockIdx.x * 8 + wave) * 16;

  if (lane < 16) {
    int e = tileBase + lane;
    if (e >= N_EDGES) e = N_EDGES - 1;
    ssrc[wave * 16 + lane] = ei[e];             // src = ei[0][e]
    sdst[wave * 16 + lane] = ei[N_EDGES + e];   // dst = ei[1][e]
  }
  __syncthreads();

  // Stage A tile: 16 edges x (x[dst](64) | x[src](64) | edge_mlp(64)) -> f16
  _Float16* A = &ldsA[wave * 16 * 192];
  for (int it = lane; it < 16 * 192; it += 32) {
    int t = it / 192, k = it - t * 192;
    int e = tileBase + t;
    if (e >= N_EDGES) e = N_EDGES - 1;
    float v;
    if (k < 64)       v = x[(long)sdst[wave * 16 + t] * 64 + k];
    else if (k < 128) v = x[(long)ssrc[wave * 16 + t] * 64 + (k - 64)];
    else              v = ea[e] * Wedge[k - 128] + bedge[k - 128];
    A[t * 192 + k] = (_Float16)v;
  }
  __syncthreads();

  v8f acc[4];
#pragma unroll
  for (int nt = 0; nt < 4; ++nt) acc[nt] = v8f_zero();

  for (int kk = 0; kk < 6; ++kk) {             // K = 192 = 6 * 32
    // A fragment: 16-bit 16x32 layout (lane-half selects K+8 / K+24 blocks)
    const _Float16* row = &A[(lane & 15) * 192 + kk * 32 + ((lane >> 4) << 3)];
    union { uint4 q[2]; v16h v; } ua;
    ua.q[0] = *(const uint4*)(row);
    ua.q[1] = *(const uint4*)(row + 16);
#pragma unroll
    for (int nt = 0; nt < 4; ++nt) {           // N = 64 = 4 * 16
      v16h b = *(const v16h*)(wpreF + ((long)(kk * 4 + nt) * 32 + lane) * 16);
      acc[nt] = wmma16(ua.v, b, acc[nt]);
    }
  }

  // Epilogue: C layout -> lane&15 = N col, vgpr r -> M = r + 8*(lane>=16)
  const int m0 = (lane >> 4) << 3;
  const int ncol = lane & 15;
#pragma unroll
  for (int nt = 0; nt < 4; ++nt) {
#pragma unroll
    for (int r = 0; r < 8; ++r) {
      int t = r + m0;
      int e = tileBase + t;
      if (e < N_EDGES) {
        int f = nt * 16 + ncol;
        float v = acc[nt][r] + bpre[f];
        int off = sdst[wave * 16 + t] * 64 + f;
        atomicAdd(&sum[off], v);
        atomicAdd(&sum2[off], v * v);
        atomicMin(&mink[off], f2ord(v));
        atomicMax(&maxk[off], f2ord(v));
      }
    }
  }
}

// ---------------------------------------------------- node GEMMs + pooling --
// Per wave: 16 nodes. Build 13F=832 feature row (x | agg | agg*amp | agg/amp)
// into LDS as f16, GEMM1 [16x832]@[832x128], GEMM2 [16x128]@[128x128],
// then atomic-scatter into per-graph pooled sums.
__global__ __launch_bounds__(64)
void k_node_gemm(const float* __restrict__ x, const int* __restrict__ batch,
                 const float* __restrict__ sum, const float* __restrict__ sum2,
                 const int* __restrict__ mink, const int* __restrict__ maxk,
                 const float* __restrict__ cnt,
                 const float* __restrict__ bpost, const float* __restrict__ blin,
                 const _Float16* __restrict__ wpostF,
                 const _Float16* __restrict__ wlinF,
                 float* __restrict__ zsum) {
  __shared__ _Float16 ldsA[2 * 16 * 832] __attribute__((aligned(16)));
  __shared__ _Float16 ldsH[2 * 16 * 128] __attribute__((aligned(16)));
  __shared__ int sb[2 * 16];

  const int lane = threadIdx.x & 31;
  const int wave = threadIdx.x >> 5;
  const int tileBase = (blockIdx.x * 2 + wave) * 16;
  const float INV_AVG_LOG = 1.0f / 2.1972245773362196f;  // 1/log(9)

  if (lane < 16) {
    int nd = tileBase + lane;
    if (nd >= N_NODES) nd = N_NODES - 1;
    sb[wave * 16 + lane] = batch[nd];
  }

  _Float16* A = &ldsA[wave * 16 * 832];
  for (int it = lane; it < 16 * 832; it += 32) {
    int t = it / 832, k = it - t * 832;
    int nd = tileBase + t;
    if (nd >= N_NODES) nd = N_NODES - 1;
    float v;
    if (k < 64) {
      v = x[(long)nd * 64 + k];
    } else {
      int j   = (k - 64) >> 6;   // 0..11
      int f   = (k - 64) & 63;
      int typ = j & 3;           // mean, min, max, std
      int sg  = j >> 2;          // *1, *amp, /amp
      float c  = cnt[nd];
      float c1 = fmaxf(c, 1.0f);
      int off  = nd * 64 + f;
      float base;
      if (typ == 0) {
        base = sum[off] / c1;
      } else if (typ == 1) {
        base = (c > 0.f) ? ord2f(mink[off]) : 0.f;
      } else if (typ == 2) {
        base = (c > 0.f) ? ord2f(maxk[off]) : 0.f;
      } else {
        float mu = sum[off] / c1;
        float m2 = sum2[off] / c1;
        base = sqrtf(fmaxf(m2 - mu * mu, 0.f) + 1e-5f);
      }
      float amp = logf(c1 + 1.f) * INV_AVG_LOG;
      v = (sg == 0) ? base : (sg == 1 ? base * amp : base / amp);
    }
    A[t * 832 + k] = (_Float16)v;
  }
  __syncthreads();

  // GEMM1: K = 832 = 26*32, N = 128 = 8*16
  v8f acc[8];
#pragma unroll
  for (int nt = 0; nt < 8; ++nt) acc[nt] = v8f_zero();
  for (int kk = 0; kk < 26; ++kk) {
    const _Float16* row = &A[(lane & 15) * 832 + kk * 32 + ((lane >> 4) << 3)];
    union { uint4 q[2]; v16h v; } ua;
    ua.q[0] = *(const uint4*)(row);
    ua.q[1] = *(const uint4*)(row + 16);
#pragma unroll
    for (int nt = 0; nt < 8; ++nt) {
      v16h b = *(const v16h*)(wpostF + ((long)(kk * 8 + nt) * 32 + lane) * 16);
      acc[nt] = wmma16(ua.v, b, acc[nt]);
    }
  }

  // h1 = acc + b_post -> LDS (f16, row-major 16x128)
  _Float16* H = &ldsH[wave * 16 * 128];
  const int m0 = (lane >> 4) << 3;
  const int ncol = lane & 15;
#pragma unroll
  for (int nt = 0; nt < 8; ++nt)
#pragma unroll
    for (int r = 0; r < 8; ++r) {
      int f = nt * 16 + ncol;
      H[(r + m0) * 128 + f] = (_Float16)(acc[nt][r] + bpost[f]);
    }
  __syncthreads();

  // GEMM2: K = 128 = 4*32, N = 128 = 8*16
  v8f acc2[8];
#pragma unroll
  for (int nt = 0; nt < 8; ++nt) acc2[nt] = v8f_zero();
  for (int kk = 0; kk < 4; ++kk) {
    const _Float16* row = &H[(lane & 15) * 128 + kk * 32 + ((lane >> 4) << 3)];
    union { uint4 q[2]; v16h v; } ua;
    ua.q[0] = *(const uint4*)(row);
    ua.q[1] = *(const uint4*)(row + 16);
#pragma unroll
    for (int nt = 0; nt < 8; ++nt) {
      v16h b = *(const v16h*)(wlinF + ((long)(kk * 8 + nt) * 32 + lane) * 16);
      acc2[nt] = wmma16(ua.v, b, acc2[nt]);
    }
  }

  // Pool into per-graph sums
#pragma unroll
  for (int nt = 0; nt < 8; ++nt)
#pragma unroll
    for (int r = 0; r < 8; ++r) {
      int t = r + m0;
      int nd = tileBase + t;
      if (nd < N_NODES) {
        int f = nt * 16 + ncol;
        atomicAdd(&zsum[sb[wave * 16 + t] * 128 + f], acc2[nt][r] + blin[f]);
      }
    }
}

// ------------------------------------------------------------- head kernel --
__device__ __forceinline__ void bn_stats64(const float* v, float& mu, float& rs) {
  float s = 0.f;
#pragma unroll 8
  for (int g = 0; g < 64; ++g) s += v[g];
  mu = s * (1.0f / 64.0f);
  float q = 0.f;
#pragma unroll 8
  for (int g = 0; g < 64; ++g) { float d = v[g] - mu; q += d * d; }
  rs = rsqrtf(q * (1.0f / 64.0f) + 1e-5f);
}

__global__ __launch_bounds__(128)
void k_head(const float* __restrict__ zsum, const float* __restrict__ gcnt,
            const float* __restrict__ g1, const float* __restrict__ be1,
            const float* __restrict__ W2, const float* __restrict__ b2,
            const float* __restrict__ g2, const float* __restrict__ be2,
            const float* __restrict__ Wr1, const float* __restrict__ br1,
            const float* __restrict__ gr1, const float* __restrict__ ber1,
            const float* __restrict__ Wr2, const float* __restrict__ br2,
            const float* __restrict__ gr2, const float* __restrict__ ber2,
            const float* __restrict__ Wout, const float* __restrict__ bout,
            float* __restrict__ out) {
  __shared__ float buf[64][132];
  const int f = threadIdx.x;   // one feature column per thread
  float col[64], res[64], acc[64];
  float mu, rs;

  // z = pooled mean; bn1 + relu
  for (int g = 0; g < 64; ++g) col[g] = zsum[g * 128 + f] / fmaxf(gcnt[g], 1.f);
  bn_stats64(col, mu, rs);
  for (int g = 0; g < 64; ++g) {
    col[g] = fmaxf((col[g] - mu) * rs * g1[f] + be1[f], 0.f);
    buf[g][f] = col[g];
  }
  __syncthreads();

  // fc2 + bn2 + relu (save residual)
  for (int g = 0; g < 64; ++g) {
    float s = b2[f];
    for (int k = 0; k < 128; ++k) s += buf[g][k] * W2[k * 128 + f];
    acc[g] = s;
  }
  bn_stats64(acc, mu, rs);
  for (int g = 0; g < 64; ++g) {
    acc[g] = fmaxf((acc[g] - mu) * rs * g2[f] + be2[f], 0.f);
    res[g] = acc[g];
  }
  __syncthreads();
  for (int g = 0; g < 64; ++g) buf[g][f] = acc[g];
  __syncthreads();

  // resnet fc1 + bn + relu
  for (int g = 0; g < 64; ++g) {
    float s = br1[f];
    for (int k = 0; k < 128; ++k) s += buf[g][k] * Wr1[k * 128 + f];
    acc[g] = s;
  }
  bn_stats64(acc, mu, rs);
  for (int g = 0; g < 64; ++g)
    acc[g] = fmaxf((acc[g] - mu) * rs * gr1[f] + ber1[f], 0.f);
  __syncthreads();
  for (int g = 0; g < 64; ++g) buf[g][f] = acc[g];
  __syncthreads();

  // resnet fc2 + bn + residual + relu
  for (int g = 0; g < 64; ++g) {
    float s = br2[f];
    for (int k = 0; k < 128; ++k) s += buf[g][k] * Wr2[k * 128 + f];
    acc[g] = s;
  }
  bn_stats64(acc, mu, rs);
  for (int g = 0; g < 64; ++g)
    acc[g] = fmaxf((acc[g] - mu) * rs * gr2[f] + ber2[f] + res[g], 0.f);
  __syncthreads();
  for (int g = 0; g < 64; ++g) buf[g][f] = acc[g];
  __syncthreads();

  // final linear to scalar per graph
  if (f < 64) {
    float s = bout[0];
    for (int k = 0; k < 128; ++k) s += buf[f][k] * Wout[k];
    out[f] = s;
  }
}

// ----------------------------------------------------------------- launch ---
extern "C" void kernel_launch(void* const* d_in, const int* in_sizes, int n_in,
                              void* d_out, int out_size, void* d_ws, size_t ws_size,
                              hipStream_t stream) {
  (void)in_sizes; (void)n_in; (void)out_size; (void)ws_size;
  const float* x     = (const float*)d_in[0];
  const int*   ei    = (const int*)d_in[1];
  const float* ea    = (const float*)d_in[2];
  const int*   batch = (const int*)d_in[3];
  const float* Wedge = (const float*)d_in[4];
  const float* bedge = (const float*)d_in[5];
  const float* Wpre  = (const float*)d_in[6];
  const float* bpre  = (const float*)d_in[7];
  const float* Wpost = (const float*)d_in[8];
  const float* bpost = (const float*)d_in[9];
  const float* Wlin  = (const float*)d_in[10];
  const float* blin  = (const float*)d_in[11];
  const float* g1    = (const float*)d_in[12];
  const float* be1   = (const float*)d_in[13];
  const float* W2    = (const float*)d_in[14];
  const float* b2    = (const float*)d_in[15];
  const float* g2    = (const float*)d_in[16];
  const float* be2   = (const float*)d_in[17];
  const float* Wr1   = (const float*)d_in[18];
  const float* br1   = (const float*)d_in[19];
  const float* gr1   = (const float*)d_in[20];
  const float* ber1  = (const float*)d_in[21];
  const float* Wr2   = (const float*)d_in[22];
  const float* br2   = (const float*)d_in[23];
  const float* gr2   = (const float*)d_in[24];
  const float* ber2  = (const float*)d_in[25];
  const float* Wout  = (const float*)d_in[26];
  const float* bout  = (const float*)d_in[27];
  float* out = (float*)d_out;

  char* ws = (char*)d_ws;
  size_t off = 0;
  auto take = [&](size_t bytes) -> char* {
    char* p = ws + off;
    off = (off + bytes + 255) & ~(size_t)255;
    return p;
  };
  const long NA = (long)N_NODES * 64;
  float*    sum    = (float*)take(NA * 4);
  float*    sum2   = (float*)take(NA * 4);
  int*      mink   = (int*)take(NA * 4);
  int*      maxk   = (int*)take(NA * 4);
  float*    cnt    = (float*)take((size_t)N_NODES * 4);
  float*    zsum   = (float*)take((size_t)N_GRAPH * 128 * 4);
  float*    gcnt   = (float*)take((size_t)N_GRAPH * 4);
  _Float16* wpreF  = (_Float16*)take((size_t)6 * 4 * 512 * 2);     // 192x64
  _Float16* wpostF = (_Float16*)take((size_t)26 * 8 * 512 * 2);    // 832x128
  _Float16* wlinF  = (_Float16*)take((size_t)4 * 8 * 512 * 2);     // 128x128

  // init aggregates (every call: harness does not re-zero ws between replays)
  k_fill_f32<<<512, 256, 0, stream>>>(sum, NA, 0.f);
  k_fill_f32<<<512, 256, 0, stream>>>(sum2, NA, 0.f);
  k_fill_i32<<<512, 256, 0, stream>>>(mink, NA, 0x7f800000);          // ord(+inf)
  k_fill_i32<<<512, 256, 0, stream>>>(maxk, NA, (int)0x807fffffu);    // ord(-inf)
  k_fill_f32<<<64, 256, 0, stream>>>(cnt, N_NODES, 0.f);
  k_fill_f32<<<4, 256, 0, stream>>>(zsum, N_GRAPH * 128, 0.f);
  k_fill_f32<<<1, 64, 0, stream>>>(gcnt, N_GRAPH, 0.f);

  // weight conversion to f16 WMMA-fragment layout
  k_convert_frag<<<48, 256, 0, stream>>>(Wpre, wpreF, 6, 4, 64);
  k_convert_frag<<<416, 256, 0, stream>>>(Wpost, wpostF, 26, 8, 128);
  k_convert_frag<<<64, 256, 0, stream>>>(Wlin, wlinF, 4, 8, 128);

  // degree / graph-size counts
  k_edge_count<<<(N_EDGES + 255) / 256, 256, 0, stream>>>(ei, cnt);
  k_graph_count<<<(N_NODES + 255) / 256, 256, 0, stream>>>(batch, gcnt);

  // edge MLP (WMMA) + scatter-reduce aggregates
  k_edge_gemm<<<N_EDGES / (16 * 8), 256, 0, stream>>>(
      x, ei, ea, Wedge, bedge, bpre, wpreF, sum, sum2, mink, maxk);

  // node post GEMMs (WMMA) + graph pooling
  int nodeTiles = (N_NODES + 15) / 16;
  k_node_gemm<<<(nodeTiles + 1) / 2, 64, 0, stream>>>(
      x, batch, sum, sum2, mink, maxk, cnt, bpost, blin, wpostF, wlinF, zsum);

  // tiny MLP head
  k_head<<<1, 128, 0, stream>>>(zsum, gcnt, g1, be1, W2, b2, g2, be2,
                                Wr1, br1, gr1, ber1, Wr2, br2, gr2, ber2,
                                Wout, bout, out);
}
